// ParallelDroplessMLP_53850299957845
// MI455X (gfx1250) — compile-verified, matrix operand order checked by reading, not run on hardware
//
#include <hip/hip_runtime.h>
#include <hip/hip_bf16.h>

// ---------------------------------------------------------------------------
// MoE dropless MLP for MI455X (gfx1250, wave32, WMMA bf16 16x16x32).
//
// Pipeline (deterministic, graph-capture safe, no atomics):
//   1) router:        stable counting sort of 8192 slots by expert
//   2) w transpose:   fp32 [R][C] -> bf16 [C][R]  (K-contiguous B operand)
//   3) gather:        xs_bf16[p][:] = bf16(x[tok_sorted[p]][:])
//   4) gemm1 + gelu:  h = gelu(xs @ w1[e])          (bf16 out)
//   5) gemm2:         ys = h @ w2[e]                (fp32 out)
//   6) combine:       out[t] = bias + sum_k w(t,k) * ys[pos(t,k)]
// ---------------------------------------------------------------------------

#define SL   2048
#define BS   2
#define HS   1024
#define FF   4096
#define NE   8
#define TOPK 2
#define T    (SL * BS)          // 4096 tokens
#define NS   (T * TOPK)         // 8192 slots

typedef __attribute__((ext_vector_type(16))) __bf16 v16bf;
typedef __attribute__((ext_vector_type(8)))  float  v8f;
typedef __attribute__((ext_vector_type(4)))  __bf16 bf4;

// ---- workspace layout (bytes) ---------------------------------------------
#define O_EOFF  ((size_t)0)                         // 9 ints
#define O_TOK   ((size_t)4096)                      // NS ints
#define O_POS   ((size_t)(4096 + 32768))            // NS ints
#define O_XS    ((size_t)131072)                    // NS*HS bf16   = 16 MB
#define O_W1T   ((size_t)33554432)                  // 32 MB: E*HS*FF bf16 = 64 MB
#define O_W2T   ((size_t)100663296)                 // 96 MB: 64 MB
#define O_H     ((size_t)167772160)                 // 160 MB: NS*FF bf16 = 64 MB
#define O_YS    ((size_t)234881024)                 // 224 MB: NS*HS f32 = 32 MB
// total: 256 MB

// ---------------------------------------------------------------------------
// 1) Router: stable counting sort by expert id. One block, 256 threads,
//    32 slots per thread. Deterministic (prefix scan, no atomics).
// ---------------------------------------------------------------------------
__global__ void moe_router_kernel(const int* __restrict__ eidx,   // [NS]
                                  int* __restrict__ tok_sorted,   // [NS]
                                  int* __restrict__ pos_of_slot,  // [NS]
                                  int* __restrict__ eoff)         // [NE+1]
{
    __shared__ int cnts[256][NE];
    __shared__ int offs[NE + 1];
    const int tid = threadIdx.x;
    const int s0  = tid * 32;

    int loc[NE];
#pragma unroll
    for (int e = 0; e < NE; ++e) loc[e] = 0;
    for (int i = 0; i < 32; ++i) loc[eidx[s0 + i]]++;
#pragma unroll
    for (int e = 0; e < NE; ++e) cnts[tid][e] = loc[e];
    __syncthreads();

    // exclusive prefix over threads, one expert per lane (tid < NE)
    __shared__ int totals[NE];
    if (tid < NE) {
        int run = 0;
        for (int t = 0; t < 256; ++t) {
            int v = cnts[t][tid];
            cnts[t][tid] = run;
            run += v;
        }
        totals[tid] = run;
    }
    __syncthreads();
    if (tid == 0) {
        int acc = 0;
        for (int e = 0; e < NE; ++e) { offs[e] = acc; acc += totals[e]; }
        offs[NE] = acc;
        for (int e = 0; e <= NE; ++e) eoff[e] = offs[e];
    }
    __syncthreads();

    int run[NE];
#pragma unroll
    for (int e = 0; e < NE; ++e) run[e] = cnts[tid][e];
    for (int i = 0; i < 32; ++i) {
        const int s = s0 + i;
        const int e = eidx[s];
        const int p = offs[e] + run[e]++;
        tok_sorted[p]  = s / TOPK;
        pos_of_slot[s] = p;
    }
}

// ---------------------------------------------------------------------------
// 2) fp32 [E][R][C] -> bf16 [E][C][R], LDS-tiled 32x32 transpose.
//    grid = (C/32, R/32, E), block = 256.
// ---------------------------------------------------------------------------
__global__ void moe_wconv_kernel(const float* __restrict__ in,
                                 __bf16* __restrict__ out,
                                 int R, int C)
{
    __shared__ float tile[32][33];
    const int e  = blockIdx.z;
    const int cb = blockIdx.x * 32;
    const int rb = blockIdx.y * 32;
    const int tx = threadIdx.x & 31;
    const int ty = threadIdx.x >> 5;     // 0..7
    const float*  ip = in  + (size_t)e * R * C;
    __bf16*       op = out + (size_t)e * R * C;

#pragma unroll
    for (int i = 0; i < 4; ++i) {
        const int r = ty + i * 8;
        tile[r][tx] = ip[(size_t)(rb + r) * C + (cb + tx)];
    }
    __syncthreads();
#pragma unroll
    for (int i = 0; i < 4; ++i) {
        const int cc = ty + i * 8;
        op[(size_t)(cb + cc) * R + (rb + tx)] = (__bf16)tile[tx][cc];
    }
}

// ---------------------------------------------------------------------------
// 3) Gather tokens into per-expert contiguous rows, fp32 -> bf16.
//    grid = NS blocks, 256 threads, 4 elems/thread.
// ---------------------------------------------------------------------------
__global__ void moe_gather_kernel(const float* __restrict__ x,       // [T][HS]
                                  const int* __restrict__ tok_sorted,
                                  __bf16* __restrict__ xs)           // [NS][HS]
{
    const int p = blockIdx.x;
    const int t = tok_sorted[p];
    const int i = threadIdx.x * 4;
    const float4 v = *(const float4*)(x + (size_t)t * HS + i);
    bf4 b;
    b.x = (__bf16)v.x; b.y = (__bf16)v.y; b.z = (__bf16)v.z; b.w = (__bf16)v.w;
    *(bf4*)(xs + (size_t)p * HS + i) = b;
}

// ---------------------------------------------------------------------------
// Grouped GEMM core: C[M_e x N] = A[M_e x K] * B[N x K]^T  (B is K-contiguous)
// 128x128 block tile, 8 waves, each wave 32x64 via 2x4 v_wmma_f32_16x16x32_bf16
// accumulators. K-step = 64 (two 32-wide WMMA sub-steps per barrier pair).
// grid = (N/128, 64, NE).
// ---------------------------------------------------------------------------
#define LDSS 72   // LDS row stride in bf16 elements (144B, keeps 16B alignment)

__device__ __forceinline__ float gelu_exact(float x) {
    return 0.5f * x * (1.0f + erff(x * 0.70710678118654752f));
}

__device__ __forceinline__ v16bf load_frag(const __bf16* base, int row, int koff) {
    union { v16bf v; uint4 q[2]; } u;
    const __bf16* r = base + row * LDSS + koff;    // koff = c*32 + (0|8) elements
    u.q[0] = *(const uint4*)(r);                   // K 0..7  (or 8..15)
    u.q[1] = *(const uint4*)(r + 16);              // K 16..23 (or 24..31)
    return u.v;
}

template <bool GELU, typename OutT>
__global__ void moe_gemm_kernel(const __bf16* __restrict__ A,      // [NS][K]
                                const __bf16* __restrict__ Bw,     // [NE][N][K]
                                OutT* __restrict__ Cp,             // [NS][N]
                                const int* __restrict__ eoff,
                                int Ktot, int N)
{
    const int e      = blockIdx.z;
    const int mstart = eoff[e] + blockIdx.y * 128;
    const int rowEnd = eoff[e + 1];
    if (mstart >= rowEnd) return;                  // block-uniform early exit
    const int n0 = blockIdx.x * 128;
    const __bf16* B = Bw + (size_t)e * N * Ktot;

    __shared__ __align__(16) __bf16 As[128 * LDSS];
    __shared__ __align__(16) __bf16 Bs[128 * LDSS];

    const int tid   = threadIdx.x;
    const int lane  = tid & 31;
    const int wave  = tid >> 5;
    const int wm    = wave & 3;                    // M group (32 rows)
    const int wn    = wave >> 2;                   // N group (64 cols)
    const int lr    = lane & 15;
    const int khalf = (lane & 16) ? 8 : 0;         // ISA 16-bit A-frag layout

    // --- staging: one loop-carried pointer per matrix per thread -----------
    // 128 rows x 64 K-elements per tile; each thread moves 64B from one row.
    const int srow  = tid >> 1;                    // 0..127
    const int shalf = (tid & 1) * 32;              // element offset in K-tile
    int garow = mstart + srow;
    garow = (garow < rowEnd) ? garow : (rowEnd - 1);   // clamp once, not per-iter
    const __bf16* aSrc = A + (size_t)garow      * Ktot + shalf;
    const __bf16* bSrc = B + (size_t)(n0 + srow) * Ktot + shalf;
    __bf16* aDst = &As[srow * LDSS + shalf];
    __bf16* bDst = &Bs[srow * LDSS + shalf];

    v8f acc[2][4];
#pragma unroll
    for (int i = 0; i < 2; ++i)
#pragma unroll
        for (int j = 0; j < 4; ++j) acc[i][j] = (v8f){};

    for (int kt = 0; kt < Ktot; kt += 64) {
        __syncthreads();                           // LDS reads of prev tile done
        {
            const uint4 a0 = ((const uint4*)aSrc)[0];
            const uint4 a1 = ((const uint4*)aSrc)[1];
            const uint4 a2 = ((const uint4*)aSrc)[2];
            const uint4 a3 = ((const uint4*)aSrc)[3];
            const uint4 b0 = ((const uint4*)bSrc)[0];
            const uint4 b1 = ((const uint4*)bSrc)[1];
            const uint4 b2 = ((const uint4*)bSrc)[2];
            const uint4 b3 = ((const uint4*)bSrc)[3];
            ((uint4*)aDst)[0] = a0; ((uint4*)aDst)[1] = a1;
            ((uint4*)aDst)[2] = a2; ((uint4*)aDst)[3] = a3;
            ((uint4*)bDst)[0] = b0; ((uint4*)bDst)[1] = b1;
            ((uint4*)bDst)[2] = b2; ((uint4*)bDst)[3] = b3;
        }
        // branchless speculative prefetch of the next weight K-tile
        __builtin_prefetch(bSrc + 64, 0, 1);
        aSrc += 64;
        bSrc += 64;
        __syncthreads();                           // tile visible to all waves

#pragma unroll
        for (int c = 0; c < 2; ++c) {
            const int ko = c * 32 + khalf;
            const v16bf a0 = load_frag(As, wm * 32 + lr,      ko);
            const v16bf a1 = load_frag(As, wm * 32 + 16 + lr, ko);
            const v16bf b0 = load_frag(Bs, wn * 64 + 0  + lr, ko);
            const v16bf b1 = load_frag(Bs, wn * 64 + 16 + lr, ko);
            const v16bf b2 = load_frag(Bs, wn * 64 + 32 + lr, ko);
            const v16bf b3 = load_frag(Bs, wn * 64 + 48 + lr, ko);

            acc[0][0] = __builtin_amdgcn_wmma_f32_16x16x32_bf16(false, a0, false, b0, (short)0, acc[0][0], false, false);
            acc[0][1] = __builtin_amdgcn_wmma_f32_16x16x32_bf16(false, a0, false, b1, (short)0, acc[0][1], false, false);
            acc[0][2] = __builtin_amdgcn_wmma_f32_16x16x32_bf16(false, a0, false, b2, (short)0, acc[0][2], false, false);
            acc[0][3] = __builtin_amdgcn_wmma_f32_16x16x32_bf16(false, a0, false, b3, (short)0, acc[0][3], false, false);
            acc[1][0] = __builtin_amdgcn_wmma_f32_16x16x32_bf16(false, a1, false, b0, (short)0, acc[1][0], false, false);
            acc[1][1] = __builtin_amdgcn_wmma_f32_16x16x32_bf16(false, a1, false, b1, (short)0, acc[1][1], false, false);
            acc[1][2] = __builtin_amdgcn_wmma_f32_16x16x32_bf16(false, a1, false, b2, (short)0, acc[1][2], false, false);
            acc[1][3] = __builtin_amdgcn_wmma_f32_16x16x32_bf16(false, a1, false, b3, (short)0, acc[1][3], false, false);
        }
    }

    // C/D layout: vgpr r, lanes 0-15 -> M=r, N=lane; lanes 16-31 -> M=r+8
    const int mh = (lane >> 4) * 8;
#pragma unroll
    for (int i = 0; i < 2; ++i) {
#pragma unroll
        for (int j = 0; j < 4; ++j) {
            const int col = n0 + wn * 64 + j * 16 + lr;
#pragma unroll
            for (int r = 0; r < 8; ++r) {
                const int gm = mstart + wm * 32 + i * 16 + mh + r;
                if (gm < rowEnd) {
                    float v = acc[i][j][r];
                    if (GELU) v = gelu_exact(v);
                    Cp[(size_t)gm * N + col] = (OutT)v;
                }
            }
        }
    }
}

// ---------------------------------------------------------------------------
// 6) Gated combine via inverse permutation (deterministic, no atomics).
//    grid = T blocks, 256 threads, 4 floats/thread.
// ---------------------------------------------------------------------------
__global__ void moe_combine_kernel(const float* __restrict__ ys,        // [NS][HS]
                                   const int* __restrict__ pos_of_slot, // [NS]
                                   const float* __restrict__ ew,        // [T][TOPK]
                                   const float* __restrict__ bias,      // [HS]
                                   float* __restrict__ out)             // [T][HS]
{
    const int t = blockIdx.x;
    const int i = threadIdx.x * 4;
    const int   p0 = pos_of_slot[t * TOPK + 0];
    const int   p1 = pos_of_slot[t * TOPK + 1];
    const float w0 = ew[t * TOPK + 0];
    const float w1 = ew[t * TOPK + 1];
    const float4 b  = *(const float4*)(bias + i);
    const float4 y0 = *(const float4*)(ys + (size_t)p0 * HS + i);
    const float4 y1 = *(const float4*)(ys + (size_t)p1 * HS + i);
    float4 o;
    o.x = b.x + w0 * y0.x + w1 * y1.x;
    o.y = b.y + w0 * y0.y + w1 * y1.y;
    o.z = b.z + w0 * y0.z + w1 * y1.z;
    o.w = b.w + w0 * y0.w + w1 * y1.w;
    *(float4*)(out + (size_t)t * HS + i) = o;
}

// ---------------------------------------------------------------------------
extern "C" void kernel_launch(void* const* d_in, const int* in_sizes, int n_in,
                              void* d_out, int out_size, void* d_ws, size_t ws_size,
                              hipStream_t stream) {
    const float* x    = (const float*)d_in[0];   // [SL][BS][HS]
    // d_in[1] = scores (unused by the reference output)
    const float* ew   = (const float*)d_in[2];   // [T][TOPK]
    const int*   eidx = (const int*)  d_in[3];   // [T][TOPK]
    const float* w1   = (const float*)d_in[4];   // [E][HS][FF]
    const float* w2   = (const float*)d_in[5];   // [E][FF][HS]
    const float* bias = (const float*)d_in[6];   // [HS]
    float*       out  = (float*)d_out;

    char* ws = (char*)d_ws;
    int*    eoff = (int*)   (ws + O_EOFF);
    int*    tok  = (int*)   (ws + O_TOK);
    int*    pos  = (int*)   (ws + O_POS);
    __bf16* xs   = (__bf16*)(ws + O_XS);
    __bf16* w1t  = (__bf16*)(ws + O_W1T);
    __bf16* w2t  = (__bf16*)(ws + O_W2T);
    __bf16* h    = (__bf16*)(ws + O_H);
    float*  ys   = (float*) (ws + O_YS);

    moe_router_kernel<<<1, 256, 0, stream>>>(eidx, tok, pos, eoff);

    moe_wconv_kernel<<<dim3(FF / 32, HS / 32, NE), 256, 0, stream>>>(w1, w1t, HS, FF);
    moe_wconv_kernel<<<dim3(HS / 32, FF / 32, NE), 256, 0, stream>>>(w2, w2t, FF, HS);

    moe_gather_kernel<<<NS, 256, 0, stream>>>(x, tok, xs);

    // GEMM1: [M_e x HS] @ [HS x FF] -> gelu -> h (bf16)
    moe_gemm_kernel<true, __bf16>
        <<<dim3(FF / 128, NS / 128, NE), 256, 0, stream>>>(xs, w1t, h, eoff, HS, FF);

    // GEMM2: [M_e x FF] @ [FF x HS] -> ys (fp32)
    moe_gemm_kernel<false, float>
        <<<dim3(HS / 128, NS / 128, NE), 256, 0, stream>>>(h, w2t, ys, eoff, FF, HS);

    moe_combine_kernel<<<T, 256, 0, stream>>>(ys, pos, ew, bias, out);
}
